// RegularSimilar_79173427134664
// MI455X (gfx1250) — matrix operand order repaired and checked by reading, not compiled, not run on hardware
//
#include <hip/hip_runtime.h>
#include <hip/hip_bf16.h>

// ---------------------------------------------------------------------------
// RegularSimilar fused pipeline for MI455X (gfx1250, wave32, WMMA).
//
// Forward math: with TAU=1e-4 the gumbel-softmax is numerically a hard
// one-hot at argmax(score + gumbel), so we never materialize the 512x100000
// score matrix: fused bf16 WMMA GEMM + streaming per-row argmax.
//
// Noise path is deliberately transcendental-free: gfx1250 tracks 16-bit XDL
// WMMA as TRANS ops (co-executes with plain VALU), so keeping the gumbel
// generator on the int/FMA VALU pipe avoids contending with the WMMA stream.
// ---------------------------------------------------------------------------

typedef __attribute__((ext_vector_type(16))) __bf16 bf16x16;
typedef __attribute__((ext_vector_type(8)))  float  f32x8;

#define BB      512
#define NN      100000
#define DD      64
#define TWOD    128
#define NTILES  (NN / 16)        // 6250 item tiles of 16
#define BTILES  (BB / 16)        // 32 row tiles of 16
#define TILE_ELEMS 1024          // 16 x 64 bf16 elements per tile (2048 B)
#define PACK_ITEMS_BYTES ((size_t)NTILES * TILE_ELEMS * 2)   // 12.8 MB
#define PACK_UIF_BYTES   ((size_t)BTILES * TILE_ELEMS * 2)   // 64 KB
#define GY      125              // N-dimension grid split
#define OUT_IDX   0
#define OUT_FEAT  BB
#define OUT_LOSS  (BB + BB * DD)
#define OUT_MSIM  (BB + BB * DD + 1)

__device__ __forceinline__ unsigned short f2bf(float f) {
  unsigned u = __float_as_uint(f);
  unsigned rnd = 0x7fffu + ((u >> 16) & 1u);   // round-to-nearest-even
  return (unsigned short)((u + rnd) >> 16);
}

// Fast log2 for positive normal floats: exponent + quadratic mantissa fit.
// Max error ~4e-3 log2-units (~3e-3 nats) -- negligible vs gumbel scale 1.28.
// Pure int/FMA VALU: no TRANS-pipe usage, no denorm fixup branches.
__device__ __forceinline__ float fast_log2(float x) {
  int xi = __float_as_int(x);
  float e = (float)((xi >> 23) - 127);
  float m = __int_as_float((xi & 0x007fffff) | 0x3f800000);   // [1,2)
  return e + fmaf(fmaf(-0.34484843f, m, 2.02466578f), m, -1.67487759f);
}

// gumbel(u) = -log(-log(u)), u in (0,1), via fast_log2 (ln x = ln2 * log2 x).
__device__ __forceinline__ float fast_gumbel(unsigned x) {
  float u = (float)(x >> 8) * (1.0f / 16777216.0f) + (0.5f / 16777216.0f);
  float E = -0.69314718f * fast_log2(u);        // Exp(1), always normal
  return   -0.69314718f * fast_log2(E);
}

// ---------------------------------------------------------------------------
// Kernel 1: convert all_items (N x 64 f32, row-major) into bf16 B-fragments.
// Per 16-item tile: frag h covers K = 32h..32h+31. Lane L holds column
// N = L%16, K = 32h + (L/16)*16 + j (j = 0..15), 16 bf16 contiguous per lane.
// ---------------------------------------------------------------------------
__global__ void __launch_bounds__(256) items_pack_kernel(
    const float* __restrict__ all_items, unsigned short* __restrict__ packed) {
  int e = blockIdx.x * blockDim.x + threadIdx.x;
  if (e >= NTILES * TILE_ELEMS) return;
  int tile = e / TILE_ELEMS;
  int r    = e % TILE_ELEMS;
  int h    = r >> 9;           // K half (0: K 0..31, 1: K 32..63)
  int s    = r & 511;
  int lane = s >> 4;
  int j    = s & 15;
  int k    = 32 * h + ((lane >> 4) << 4) + j;
  int item = tile * 16 + (lane & 15);
  packed[e] = f2bf(all_items[(size_t)item * DD + k]);
}

// ---------------------------------------------------------------------------
// Kernel 2: uif = union_feature @ W^T + b, packed into WMMA A-fragment
// layout, this file's contiguous K order: lane L<16 -> M=L, K in {0..15};
// lane>=16 -> M=L-16, K in {16..31} (matches items B-fragment K order).
// ---------------------------------------------------------------------------
__global__ void __launch_bounds__(256) uif_pack_kernel(
    const float* __restrict__ uf, const float* __restrict__ W,
    const float* __restrict__ bias, unsigned short* __restrict__ packedUif) {
  int e = blockIdx.x * blockDim.x + threadIdx.x;
  if (e >= BTILES * TILE_ELEMS) return;
  int tile = e / TILE_ELEMS;
  int r    = e % TILE_ELEMS;
  int h    = r >> 9;
  int s    = r & 511;
  int lane = s >> 4;
  int j    = s & 15;
  int klo  = (lane >= 16) ? 8 : 0;
  int k    = 32 * h + ((lane >> 4) << 4) + j;
  int row  = tile * 16 + (lane & 15);
  const float* u = uf + (size_t)row * TWOD;
  const float* w = W + (size_t)k * TWOD;
  float acc = bias[k];
  #pragma unroll 8
  for (int q = 0; q < TWOD; ++q) acc = fmaf(u[q], w[q], acc);
  packedUif[e] = f2bf(acc);
}

// ---------------------------------------------------------------------------
// Kernel 3: zero the per-row argmax accumulators (packed u64 keys).
// ---------------------------------------------------------------------------
__global__ void init_best_kernel(unsigned long long* __restrict__ rowBest) {
  rowBest[threadIdx.x] = 0ull;
}

// ---------------------------------------------------------------------------
// Kernel 4: fused WMMA GEMM (score tile) + gumbel + streaming argmax.
// Grid: (BTILES, GY); 256 threads = 8 waves. Each wave walks item tiles
// strided by 8, 2 chained v_wmma_f32_16x16x32_bf16 per tile (K = 64).
// C layout: VGPR r -> row rowBase + r, lanes L: N = L%16, rowBase += 8*(L/16).
// ---------------------------------------------------------------------------
__global__ void __launch_bounds__(256) gemm_argmax_kernel(
    const unsigned short* __restrict__ packedItems,
    const unsigned short* __restrict__ packedUif,
    unsigned long long* __restrict__ rowBest, int tilesPerY) {
  __shared__ float sV[8][16];
  __shared__ int   sI[8][16];

  const int lane = threadIdx.x & 31;
  const int wave = threadIdx.x >> 5;
  const int bt   = blockIdx.x;
  const int t0   = blockIdx.y * tilesPerY;
  int tEnd = t0 + tilesPerY;
  if (tEnd > NTILES) tEnd = NTILES;

  // A fragments (uif rows bt*16..bt*16+15), resident for the whole scan.
  const bf16x16 a0 = *reinterpret_cast<const bf16x16*>(
      packedUif + (size_t)bt * TILE_ELEMS + 0 * 512 + lane * 16);
  const bf16x16 a1 = *reinterpret_cast<const bf16x16*>(
      packedUif + (size_t)bt * TILE_ELEMS + 1 * 512 + lane * 16);

  const int nloc    = lane & 15;
  const int rowBase = bt * 16 + ((lane >> 4) << 3);

  float best[8];
  int   bidx[8];
  #pragma unroll
  for (int r = 0; r < 8; ++r) { best[r] = -3.4e38f; bidx[r] = 0; }

  for (int t = t0 + wave; t < tEnd; t += 8) {
    const unsigned short* bp = packedItems + (size_t)t * TILE_ELEMS;
    bf16x16 b0 = *reinterpret_cast<const bf16x16*>(bp + lane * 16);
    bf16x16 b1 = *reinterpret_cast<const bf16x16*>(bp + 512 + lane * 16);
    if (t + 8 < tEnd)  // gfx1250 global_prefetch_b8 on next streamed tile
      __builtin_prefetch(packedItems + (size_t)(t + 8) * TILE_ELEMS + lane * 16, 0, 1);

    f32x8 c = {};
    c = __builtin_amdgcn_wmma_f32_16x16x32_bf16(false, a0, false, b0,
                                                (short)0, c, false, false);
    c = __builtin_amdgcn_wmma_f32_16x16x32_bf16(false, a1, false, b1,
                                                (short)0, c, false, false);

    const int item = t * 16 + nloc;
    // One hash per (lane, tile); 8 per-row uniforms via 2-op LCG steps.
    unsigned x = ((unsigned)item + 0x9E3779B9u) * 0x85EBCA6Bu ^
                 ((unsigned)rowBase * 0xC2B2AE35u);
    x ^= x >> 16; x *= 0x7FEB352Du; x ^= x >> 15;
    #pragma unroll
    for (int r = 0; r < 8; ++r) {
      x = x * 1664525u + 1013904223u;
      float v = c[r] + fast_gumbel(x);
      if (v > best[r]) { best[r] = v; bidx[r] = item; }
    }
  }

  // Reduce across the 16 lanes holding the same rows (different items).
  #pragma unroll
  for (int r = 0; r < 8; ++r) {
    float v = best[r];
    int   i = bidx[r];
    #pragma unroll
    for (int off = 8; off >= 1; off >>= 1) {
      float ov = __shfl_xor(v, off, 32);
      int   oi = __shfl_xor(i, off, 32);
      if (ov > v || (ov == v && oi < i)) { v = ov; i = oi; }
    }
    if (nloc == 0) {
      int slot = ((lane >> 4) << 3) + r;   // local row 0..15
      sV[wave][slot] = v;
      sI[wave][slot] = i;
    }
  }
  __syncthreads();

  // Cross-wave reduce, then one packed u64 atomicMax per row per block.
  if (threadIdx.x < 16) {
    float v = sV[0][threadIdx.x];
    int   i = sI[0][threadIdx.x];
    #pragma unroll
    for (int wv = 1; wv < 8; ++wv) {
      float ov = sV[wv][threadIdx.x];
      int   oi = sI[wv][threadIdx.x];
      if (ov > v || (ov == v && oi < i)) { v = ov; i = oi; }
    }
    unsigned u = __float_as_uint(v);
    unsigned skey = (u & 0x80000000u) ? ~u : (u | 0x80000000u);
    unsigned long long key =
        ((unsigned long long)skey << 32) | (unsigned)i;
    atomicMax(&rowBest[bt * 16 + threadIdx.x], key);
  }
}

// ---------------------------------------------------------------------------
// Kernel 5: finalize — gather winning item features, cosine similarity,
// scalar means. One 512-thread block, LDS tree reduction.
// ---------------------------------------------------------------------------
__global__ void __launch_bounds__(512) finalize_kernel(
    const unsigned long long* __restrict__ rowBest,
    const int* __restrict__ need_replace,
    const float* __restrict__ all_items,
    float* __restrict__ out) {
  __shared__ float sLoss[512];
  __shared__ float sSim[512];
  const int bi = threadIdx.x;

  unsigned idx = (unsigned)(rowBest[bi] & 0xffffffffu);
  out[OUT_IDX + bi] = (float)idx;

  const float* feat = all_items + (size_t)idx * DD;
  const int itemId  = need_replace[bi * 2 + 1];
  const float* emb  = all_items + (size_t)itemId * DD;

  float dot = 0.f, na = 0.f, nb = 0.f;
  #pragma unroll 8
  for (int k = 0; k < DD; ++k) {
    float f = feat[k];
    float e = emb[k];
    out[OUT_FEAT + bi * DD + k] = f;
    dot = fmaf(e, f, dot);
    na  = fmaf(e, e, na);
    nb  = fmaf(f, f, nb);
  }
  float denom = fmaxf(sqrtf(na) * sqrtf(nb), 1e-6f);
  float sim = (dot / denom + 1.0f) * 0.5f;
  float d = sim - 0.5f;              // SIM_RATIO label
  sLoss[bi] = d * d;
  sSim[bi]  = sim;
  __syncthreads();
  for (int s = 256; s > 0; s >>= 1) {
    if (bi < s) { sLoss[bi] += sLoss[bi + s]; sSim[bi] += sSim[bi + s]; }
    __syncthreads();
  }
  if (bi == 0) {
    out[OUT_LOSS] = sLoss[0] * (1.0f / (float)BB);
    out[OUT_MSIM] = sSim[0] * (1.0f / (float)BB);
  }
}

// ---------------------------------------------------------------------------
extern "C" void kernel_launch(void* const* d_in, const int* in_sizes, int n_in,
                              void* d_out, int out_size, void* d_ws, size_t ws_size,
                              hipStream_t stream) {
  (void)in_sizes; (void)n_in; (void)out_size; (void)ws_size;

  const int*   need_replace  = (const int*)d_in[0];
  const float* union_feature = (const float*)d_in[1];
  const float* all_items     = (const float*)d_in[2];
  const float* W             = (const float*)d_in[3];
  const float* bvec          = (const float*)d_in[4];
  float*       out           = (float*)d_out;

  unsigned char* ws = (unsigned char*)d_ws;
  unsigned short*      packedItems = (unsigned short*)ws;
  unsigned short*      packedUif   = (unsigned short*)(ws + PACK_ITEMS_BYTES);
  unsigned long long*  rowBest     =
      (unsigned long long*)(ws + PACK_ITEMS_BYTES + PACK_UIF_BYTES);

  items_pack_kernel<<<(NTILES * TILE_ELEMS + 255) / 256, 256, 0, stream>>>(
      all_items, packedItems);
  uif_pack_kernel<<<(BTILES * TILE_ELEMS + 255) / 256, 256, 0, stream>>>(
      union_feature, W, bvec, packedUif);
  init_best_kernel<<<1, BB, 0, stream>>>(rowBest);

  const int tilesPerY = (NTILES + GY - 1) / GY;   // 50
  dim3 grid(BTILES, GY);
  gemm_argmax_kernel<<<grid, 256, 0, stream>>>(packedItems, packedUif,
                                               rowBest, tilesPerY);

  finalize_kernel<<<1, BB, 0, stream>>>(rowBest, need_replace, all_items, out);
}
